// GlobalDynamicDomainFilters_19542101196962
// MI455X (gfx1250) — matrix operand order repaired and checked by reading, not compiled, not run on hardware
//
#include <hip/hip_runtime.h>

// MI455X / gfx1250 (wave32, WMMA). Memory-bound problem (~1.36 GB moved vs ~41 GF):
// read x once (mean partials fused into the f kernel), keep f in bf16 ws,
// fuse [3x3 dynamic conv as GEMM] + [1x1 fuse GEMM] in one kernel via an LDS
// o-tile handoff. All matrix math via v_wmma_f32_16x16x32_bf16.
// Im2col K-order is tap-major (k = tau*32 + i) so every WMMA B fragment is a
// single contiguous 32-byte LDS load (no scalar gather, no runtime div/mod).

typedef __attribute__((ext_vector_type(16))) __bf16 v16bf;
typedef __attribute__((ext_vector_type(8)))  __bf16 v8bf;
typedef __attribute__((ext_vector_type(8)))  float  v8f;

#define DEV __device__ __forceinline__

constexpr int N_  = 32;    // batch
constexpr int C_  = 128;   // input channels
constexpr int H_  = 192;
constexpr int W_  = 192;
constexpr int CM  = 32;    // mid channels (Cm)
constexpr int K2  = 9;     // 3x3 taps
constexpr int D_  = 4;     // domains
constexpr int BD  = 8;     // batch per domain
constexpr int PL  = 128;   // planes (fuse out)
constexpr int WB  = 3;     // 64-pixel tiles per row (192/64)
constexpr int NBLK = H_ * WB;   // partial-sum blocks per sample = 576
constexpr int KC  = K2 * CM;    // im2col K = 288 = 9 WMMA k-steps of 32 (tap-major)

// ---- WMMA fragment helpers (A/B in contiguous k order k = 16*(L>=16) + s, same for both) ----

// A fragment (16x32 bf16, MxK): lane holds M = L&15;
// halves 0..7 -> K = kbase+0..7, halves 8..15 -> K = kbase+8..15, kbase = ks*32 + 16*(L>=16).
DEV v16bf load_a(const __bf16* row, int kbase) {
  v8bf lo = *(const v8bf*)(row + kbase);
  v8bf hi = *(const v8bf*)(row + kbase + 8);
  v16bf a;
#pragma unroll
  for (int i = 0; i < 8; ++i) { a[i] = lo[i]; a[8 + i] = hi[i]; }
  return a;
}

DEV v8f wmma_bf16(v16bf a, v16bf b, v8f c) {
  return __builtin_amdgcn_wmma_f32_16x16x32_bf16(false, a, false, b, (short)0, c,
                                                 false, false);
}

DEV float relu(float v) { return v > 0.f ? v : 0.f; }

// =====================================================================
// Kernel 1: f = relu(conv1x1(x)) in bf16, plus per-block channel sums
// (for the global mean) written deterministically to a partial buffer.
// Block = 128 threads (4 waves), covers (n, h, 64 pixels).
// =====================================================================
__global__ __launch_bounds__(128) void k_f_part(
    const float* __restrict__ x, const float* __restrict__ conv_w,
    const float* __restrict__ conv_b, __bf16* __restrict__ f_bf,
    float* __restrict__ partial) {
  __shared__ alignas(64) __bf16 x_t[64][C_];   // [pixel][K=c] contiguous B loads
  __shared__ alignas(64) __bf16 cw[CM][C_];    // conv_w bf16
  __shared__ float cb[CM];

  const int wb = blockIdx.x, h = blockIdx.y, n = blockIdx.z;
  const int wbase = wb * 64;
  const int t = threadIdx.x;  // 0..127 == channel

  // One thread per input channel: stream 64 contiguous floats, accumulate the
  // row-segment sum (deterministic partial), transpose-store bf16 to LDS.
  {
    const float* xp = x + (((size_t)(n * C_ + t) * H_ + h) * W_ + wbase);
    float s = 0.f;
#pragma unroll 4
    for (int j = 0; j < 64; ++j) {
      float v = xp[j];
      s += v;
      x_t[j][t] = (__bf16)v;
    }
    partial[(size_t)(n * C_ + t) * NBLK + (h * WB + wb)] = s;
  }
  for (int idx = t; idx < CM * C_; idx += 128)
    cw[idx / C_][idx % C_] = (__bf16)conv_w[idx];
  if (t < CM) cb[t] = conv_b[t];
  __syncthreads();

  const int lane = t & 31, wave = t >> 5;
  const int m = lane & 15, hi16 = lane >> 4;
  const int koff16 = hi16 * 16, koffA = hi16 * 16;
  const int pix = wave * 16 + m;

  v8f acc0 = {}, acc1 = {};
#pragma unroll
  for (int ks = 0; ks < 4; ++ks) {  // K = 128 = 4 x 32
    v16bf b = *(const v16bf*)&x_t[pix][ks * 32 + koff16];  // contiguous 32B
    v16bf a0 = load_a(&cw[m][0], ks * 32 + koffA);
    v16bf a1 = load_a(&cw[16 + m][0], ks * 32 + koffA);
    acc0 = wmma_bf16(a0, b, acc0);
    acc1 = wmma_bf16(a1, b, acc1);
  }
  // C/D layout: VGPR r -> M = r + 8*(L>=16), N = L&15
#pragma unroll
  for (int r = 0; r < 8; ++r) {
    int c0 = r + 8 * hi16;
    float v0 = relu(acc0[r] + cb[c0]);
    float v1 = relu(acc1[r] + cb[16 + c0]);
    f_bf[(((size_t)(n * CM + c0) * H_ + h) * W_) + wbase + pix] = (__bf16)v0;
    f_bf[(((size_t)(n * CM + 16 + c0) * H_ + h) * W_) + wbase + pix] = (__bf16)v1;
  }
}

// =====================================================================
// Kernel 2: deterministic reduce of 576 partials per (n,c) -> mean.
// =====================================================================
__global__ __launch_bounds__(256) void k_mean(const float* __restrict__ partial,
                                              float* __restrict__ mean) {
  int tid = blockIdx.x * 256 + threadIdx.x;  // 0..4095 == (n,c)
  const float* p = partial + (size_t)tid * NBLK;
  float s = 0.f;
  for (int j = 0; j < NBLK; ++j) s += p[j];
  mean[tid] = s * (1.0f / (H_ * W_));
}

// =====================================================================
// Kernel 3: tiny prep — g, per-domain gbar, dynamic 3x3 weights (bf16,
// TAP-MAJOR im2col K-order k = tau*32 + i), bf16 fuse weights, folded bias.
// =====================================================================
__global__ __launch_bounds__(256) void k_prep(
    const float* __restrict__ mean, const float* __restrict__ conv_w,
    const float* __restrict__ conv_b, const float* __restrict__ convk_w,
    const float* __restrict__ convk_b, const float* __restrict__ kern_w,
    const float* __restrict__ kern_b, const float* __restrict__ adap_b,
    const float* __restrict__ fuse_w, const float* __restrict__ fuse_b,
    __bf16* __restrict__ wts_bf, __bf16* __restrict__ fuse_bf,
    float* __restrict__ fb2) {
  __shared__ float g_s[N_][CM];
  __shared__ float gbar[D_][CM];
  const int t = threadIdx.x;

  for (int idx = t; idx < N_ * CM; idx += 256) {
    int n = idx / CM, o = idx % CM;
    float s = conv_b[o];
    for (int c = 0; c < C_; ++c) s += mean[n * C_ + c] * conv_w[o * C_ + c];
    g_s[n][o] = relu(s);
  }
  __syncthreads();
  for (int idx = t; idx < D_ * CM; idx += 256) {
    int dd = idx / CM, i = idx % CM;
    float s = 0.f;
    for (int b = 0; b < BD; ++b) s += g_s[dd * BD + b][i];
    gbar[dd][i] = s * (1.0f / BD);
  }
  __syncthreads();
  // wts[d, o, tau*32 + i] = relu((gbar[d,i]*kern_w[tau] + kern_b[tau])*convk_w[o] + convk_b[o])
  for (int idx = t; idx < D_ * CM * KC; idx += 256) {
    int dd = idx / (CM * KC), rem = idx % (CM * KC);
    int o = rem / KC, k = rem % KC;
    int tau = k / CM, i = k % CM;  // tap-major
    float v = (gbar[dd][i] * kern_w[tau] + kern_b[tau]) * convk_w[o] + convk_b[o];
    wts_bf[idx] = (__bf16)relu(v);
  }
  for (int idx = t; idx < PL * CM; idx += 256) fuse_bf[idx] = (__bf16)fuse_w[idx];
  // fb2[p] = fuse_w[p,:] @ adap_b + fuse_b[p]   (adap_b folded through fuse)
  for (int idx = t; idx < PL; idx += 256) {
    float s = fuse_b[idx];
    for (int c = 0; c < CM; ++c) s += fuse_w[idx * CM + c] * adap_b[c];
    fb2[idx] = s;
  }
}

// =====================================================================
// Kernel 4: fused [3x3 dynamic conv Cm->Cm as GEMM 32x288 tap-major] +
// [fuse GEMM 128x32] per 16-pixel tile. o never leaves the WGP
// (regs -> LDS -> WMMA B). Block = 128 threads (4 waves) = (n, h, 64 px).
// K-step ks == tap tau: dy = ks/3, dx = ks%3 are compile-time constants,
// and the B fragment is one contiguous 32B load from fpad_t[dy][p+dx][..].
// =====================================================================
__global__ __launch_bounds__(128) void k_main(
    const __bf16* __restrict__ f_bf, const __bf16* __restrict__ wts_bf,
    const __bf16* __restrict__ fuse_bf, const float* __restrict__ fb2,
    float* __restrict__ y) {
  __shared__ alignas(64) __bf16 fpad_t[3][66][CM];  // [dy][padded col][channel i]
  __shared__ alignas(64) __bf16 wlds[CM][KC];       // this domain's tap-major weights
  __shared__ alignas(64) __bf16 flds[PL][CM];       // fuse weights
  __shared__ alignas(64) __bf16 o_t[4][16][CM];     // o tile: [wave][pixel][K=c]

  const int wb = blockIdx.x, h = blockIdx.y, n = blockIdx.z;
  const int d = n >> 3;  // BD = 8
  const int wbase = wb * 64;
  const int t = threadIdx.x;

  // f patch with zero halo, transposed to channel-innermost on LDS store so
  // global reads stay contiguous per (i,row) while B fragments are contiguous.
  for (int idx = t; idx < CM * 3 * 66; idx += 128) {
    int i = idx / 198, rem = idx % 198, r = rem / 66, cidx = rem % 66;
    int gh = h - 1 + r, gw = wbase - 1 + cidx;
    __bf16 v = (__bf16)0.f;
    if ((unsigned)gh < (unsigned)H_ && (unsigned)gw < (unsigned)W_)
      v = f_bf[(((size_t)(n * CM + i) * H_ + gh) * W_) + gw];
    fpad_t[r][cidx][i] = v;
  }
  for (int idx = t; idx < CM * KC; idx += 128)
    wlds[idx / KC][idx % KC] = wts_bf[(size_t)d * CM * KC + idx];
  for (int idx = t; idx < PL * CM; idx += 128)
    flds[idx / CM][idx % CM] = fuse_bf[idx];
  __syncthreads();

  const int lane = t & 31, wave = t >> 5;
  const int m = lane & 15, hi16 = lane >> 4;
  const int koff16 = hi16 * 16, koffA = hi16 * 16;
  const int p = wave * 16 + m;  // pixel within the 64-px tile

  // ---- 3x3 conv as A[32x288] x B[288x16], 9 tap k-steps, 2 M-tiles ----
  v8f acc0 = {}, acc1 = {};
#pragma unroll
  for (int ks = 0; ks < 9; ++ks) {
    const int dy = ks / 3, dx = ks % 3;  // compile-time after unroll
    v16bf b = *(const v16bf*)&fpad_t[dy][p + dx][koff16];  // contiguous 32B
    v16bf a0 = load_a(&wlds[m][0], ks * 32 + koffA);
    v16bf a1 = load_a(&wlds[16 + m][0], ks * 32 + koffA);
    acc0 = wmma_bf16(a0, b, acc0);
    acc1 = wmma_bf16(a1, b, acc1);
  }

  // ---- hand o (32ch x 16px) to the fuse GEMM through LDS, transposed ----
#pragma unroll
  for (int r = 0; r < 8; ++r) {
    int c0 = r + 8 * hi16;
    o_t[wave][m][c0]      = (__bf16)acc0[r];
    o_t[wave][m][16 + c0] = (__bf16)acc1[r];
  }
  __syncthreads();

  v16bf bo = *(const v16bf*)&o_t[wave][m][koff16];  // contiguous 32B B fragment
  const int wcol = wbase + p;
#pragma unroll
  for (int mt = 0; mt < 8; ++mt) {  // 128 output planes = 8 M-tiles
    v8f acc = {};
    v16bf a = load_a(&flds[mt * 16 + m][0], koffA);
    acc = wmma_bf16(a, bo, acc);
#pragma unroll
    for (int r = 0; r < 8; ++r) {
      int ch = mt * 16 + r + 8 * hi16;
      y[(((size_t)(n * PL + ch) * H_ + h) * W_) + wcol] = acc[r] + fb2[ch];
    }
  }
}

// =====================================================================
extern "C" void kernel_launch(void* const* d_in, const int* in_sizes, int n_in,
                              void* d_out, int out_size, void* d_ws,
                              size_t ws_size, hipStream_t stream) {
  (void)in_sizes; (void)n_in; (void)out_size; (void)ws_size;
  const float* x       = (const float*)d_in[0];
  const float* conv_w  = (const float*)d_in[1];
  const float* conv_b  = (const float*)d_in[2];
  const float* convk_w = (const float*)d_in[3];
  const float* convk_b = (const float*)d_in[4];
  const float* kern_w  = (const float*)d_in[5];
  const float* kern_b  = (const float*)d_in[6];
  const float* adap_b  = (const float*)d_in[7];
  const float* fuse_w  = (const float*)d_in[8];
  const float* fuse_b  = (const float*)d_in[9];
  float* y = (float*)d_out;

  char* ws = (char*)d_ws;
  size_t off = 0;
  __bf16* f_bf    = (__bf16*)(ws + off); off += (size_t)N_ * CM * H_ * W_ * 2;  // 75.5 MB
  float*  partial = (float*)(ws + off);  off += (size_t)N_ * C_ * NBLK * 4;     // 9.4 MB
  float*  mean    = (float*)(ws + off);  off += (size_t)N_ * C_ * 4;
  __bf16* wts_bf  = (__bf16*)(ws + off); off += (size_t)D_ * CM * KC * 2;
  __bf16* fuse_bf = (__bf16*)(ws + off); off += (size_t)PL * CM * 2;
  float*  fb2     = (float*)(ws + off);  off += (size_t)PL * 4;

  dim3 grid(WB, H_, N_);  // (64-px tile, row, sample)
  k_f_part<<<grid, 128, 0, stream>>>(x, conv_w, conv_b, f_bf, partial);
  k_mean<<<(N_ * C_) / 256, 256, 0, stream>>>(partial, mean);
  k_prep<<<1, 256, 0, stream>>>(mean, conv_w, conv_b, convk_w, convk_b, kern_w,
                                kern_b, adap_b, fuse_w, fuse_b, wts_bf, fuse_bf,
                                fb2);
  k_main<<<grid, 128, 0, stream>>>(f_bf, wts_bf, fuse_bf, fb2, y);
}
